// Convnet_81862076661945
// MI455X (gfx1250) — matrix operand order repaired and checked by reading, not compiled, not run on hardware
//
#include <hip/hip_runtime.h>

// ---------------------------------------------------------------------------
// Problem constants (from the reference)
// ---------------------------------------------------------------------------
#define N_SECTIONS 8
#define N_CHANNELS 128
#define KT 32
#define KF 16
#define LPOST 400
#define SEC_DIST 400
#define N_T 3231
#define FREQ 256
#define FP_OUT 15              // pooled frequency bands actually used (f=240 dropped)
#define THR 15.0f

// tiling
#define TSPAN 100              // time rows per block
#define TCHUNKS 4              // 4 * 100 = 400
#define XT_ROWS (TSPAN + KT - 1)   // 131
#define XT_STRIDE 34               // padded row stride in halfs (TDM: 16dw + 1dw pad)
#define PT_STRIDE 32               // pair-table row stride in dwords

#define XH_ELEMS (N_T * FREQ)                        // 827136
#define WH_ELEMS (N_SECTIONS * N_CHANNELS * KT * KF) // 524288
#define POOL_ELEMS (N_SECTIONS * 8 * FP_OUT)         // 960 (16-bit masks per ctile)

typedef __attribute__((ext_vector_type(16))) _Float16 v16h;
typedef __attribute__((ext_vector_type(8)))  float    v8f;
typedef __attribute__((ext_vector_type(4)))  unsigned u32x4;
typedef __attribute__((ext_vector_type(8)))  int      i32x8;
typedef __attribute__((ext_vector_type(4)))  int      i32x4;

union HFrag { u32x4 u[2]; v16h h; };
union HPack { _Float16 h[2]; unsigned u; };

// ---------------------------------------------------------------------------
// Kernel 1: f32 -> f16 conversion of X and W into scratch; zero pool buffer
// ---------------------------------------------------------------------------
__global__ void cvt_kernel(const float* __restrict__ X, const float* __restrict__ W,
                           _Float16* __restrict__ Xh, _Float16* __restrict__ Wh,
                           unsigned* __restrict__ pool) {
  int i = blockIdx.x * 256 + threadIdx.x;
  if (i < XH_ELEMS) Xh[i] = (_Float16)X[i];
  if (i < WH_ELEMS) Wh[i] = (_Float16)W[i];
  if (i < POOL_ELEMS) pool[i] = 0u;
}

// ---------------------------------------------------------------------------
// Kernel 2: per (section, band, t-chunk): 16x16x32 f16 WMMA GEMM over K=512,
// threshold >= 15, OR-pool over time & band, atomicOr into pool masks.
//
// Xt staging uses the Tensor Data Mover (tensor_load_to_lds, 6-arg builtin):
// one descriptor DMAs the 131x32-half tile, with D# pad fields producing the
// 34-half LDS row pitch (pad 1 DWORD every 16 DWORDs). Completion via
// s_wait_tensorcnt. Fallback: vector-load staging loop.
//
// Fragment k order used by this code (wave32, same for A and B), K ordered k = kt*16 + kf:
//   A 16x32 f16 : lane L (m=L&15, p=L>>4) halfs j=0..15 -> K = p*16 + j
//   B 32x16 f16 : lane L (n=L&15, p=L>>4) halfs j=0..15 -> K = p*16 + j
// ---------------------------------------------------------------------------
__global__ void __launch_bounds__(256)
conv_spike_kernel(const _Float16* __restrict__ Xh, const _Float16* __restrict__ Wh,
                  unsigned* __restrict__ pool) {
  __shared__ __align__(16) _Float16 Xt[XT_ROWS * XT_STRIDE];   //  8.9 KB staging
  __shared__ __align__(16) unsigned  Pt[XT_ROWS * PT_STRIDE];  // 16.4 KB pair table
  __shared__ __align__(32) _Float16 Bf[2][16 * 512];           // 32.0 KB B fragments

  const int tid  = threadIdx.x;
  const int s    = blockIdx.z;          // section
  const int fp   = blockIdx.y;          // pooled frequency band
  const int t0   = blockIdx.x * TSPAN;  // time-chunk start
  const int wave = tid >> 5;
  const int lane = tid & 31;
  const int m    = lane & 15;
  const int p    = lane >> 4;
  const int sstart = s * SEC_DIST;

  // ---- stage X tile: rows [t0, t0+130], cols [fp*16, fp*16+31] -> LDS -----
#if __has_builtin(__builtin_amdgcn_tensor_load_to_lds) && __has_builtin(__builtin_amdgcn_s_wait_tensorcnt)
  if (tid < 32) {   // one wave issues the TDM descriptor (EXEC ignored by TDM)
    unsigned lds_base = (unsigned)(size_t)(&Xt[0]);   // flat addr[31:0] = LDS offset
    unsigned long long gaddr =
        (unsigned long long)(size_t)(Xh + (sstart + t0) * FREQ + fp * 16);
    u32x4 g0;
    g0[0] = 1u;                                        // count=1 (valid user D#)
    g0[1] = lds_base;                                  // lds_addr
    g0[2] = (unsigned)gaddr;                           // global_addr[31:0]
    g0[3] = ((unsigned)(gaddr >> 32) & 0x01FFFFFFu)    // global_addr[56:32]
            | (2u << 30);                              // type=2 ("image")
    i32x8 g1;
    g1[0] = (1 << 16)      // data_size = 1 -> 2-byte elements; workgroup_mask=0
          | (1 << 20)      // pad_enable
          | (3 << 22);     // pad_interval: every 16 DWORDs (= 32-half tile row)
                           // pad_amount = 0 -> 1 DWORD  => 34-half row pitch
    g1[1] = (int)((FREQ & 0xFFFF) << 16);   // tensor_dim0 = 256  (bits 79:48)
    g1[2] = (int)((N_T & 0xFFFF) << 16);    // tensor_dim1 = 3231 (bits 111:80)
    g1[3] = (int)(32 << 16);                // tile_dim0 = 32     (bits 127:112)
    g1[4] = 131;                            // tile_dim1 = 131; tile_dim2 = 0
    g1[5] = FREQ;                           // tensor_dim0_stride = 256 (bits 207:160)
    g1[6] = 0;                              // stride hi / tensor_dim1_stride lo
    g1[7] = 0;
    i32x4 gz4 = {0, 0, 0, 0};               // groups 2/3: 2-D tile, unused dims
    i32x8 gz8 = {0, 0, 0, 0, 0, 0, 0, 0};
    __builtin_amdgcn_tensor_load_to_lds(g0, g1, gz4, gz4, gz8, 0);
    __builtin_amdgcn_s_wait_tensorcnt(0);
  }
#else
  for (int idx = tid; idx < XT_ROWS * 16; idx += 256) {
    int r  = idx >> 4;
    int cp = idx & 15;
    const unsigned* src = (const unsigned*)(Xh + (sstart + t0 + r) * FREQ + fp * 16);
    *(unsigned*)&Xt[r * XT_STRIDE + cp * 2] = src[cp];
  }
#endif

  // ---- preload A fragments (this wave's 16 channels, full K=512) ----------
  v16h afrag[16];
  {
    const _Float16* wb = Wh + ((s * N_CHANNELS + wave * 16 + m) * (KT * KF));
#pragma unroll
    for (int kc = 0; kc < 16; ++kc) {
      HFrag f;
      f.u[0] = *(const u32x4*)(wb + (2 * kc + p) * 16);      // kt = 2*kc+p, kf = 0..7
      f.u[1] = *(const u32x4*)(wb + (2 * kc + p) * 16 + 8);  // kt = 2*kc+p, kf = 8..15
      afrag[kc] = f.h;
    }
  }
  __syncthreads();

  // ---- one-time pair table: Pt[r][c] = pack(Xt[r][c], Xt[r][c+1]) ---------
  for (int idx = tid; idx < XT_ROWS * PT_STRIDE; idx += 256) {
    int r = idx >> 5;
    int c = idx & 31;
    if (c < 31) {
      HPack pk;
      pk.h[0] = Xt[r * XT_STRIDE + c];
      pk.h[1] = Xt[r * XT_STRIDE + c + 1];
      Pt[r * PT_STRIDE + c] = pk.u;
    }
  }
  __syncthreads();

  // Producer mapping (b128-granular): thread -> kcb = tid>>6 (kc = kcb*4+it),
  // Ld = (tid>>1)&31 (fragment lane slot), jpq = tid&1 (which dword quad).
  // Fragment dword d = kc*256 + Ld*8 + jpq*4 + u  <=  Pt[(tt+2*kc+pL)*32 + nL + 8*jpq + 2u]
  const int kcb = tid >> 6;
  const int Ld  = (tid >> 1) & 31;
  const int jpq = tid & 1;
  const int pL2 = Ld >> 4;
  const int nL2 = Ld & 15;

  auto produce = [&](int tt, int buf) {
    u32x4 vals[4];
    int rd = (tt + 8 * kcb + pL2) * PT_STRIDE + nL2 + 8 * jpq;
#pragma unroll
    for (int it = 0; it < 4; ++it) {           // gather first (16 b32 loads)
      u32x4 v;
      v.x = Pt[rd];
      v.y = Pt[rd + 2];
      v.z = Pt[rd + 4];
      v.w = Pt[rd + 6];
      vals[it] = v;
      rd += 2 * PT_STRIDE;                     // kc+1 -> +2 rows
    }
    u32x4* dst = (u32x4*)((unsigned*)Bf[buf] + kcb * 1024 + Ld * 8 + jpq * 4);
#pragma unroll
    for (int it = 0; it < 4; ++it)             // then 4 b128 stores
      dst[it * 64] = vals[it];
  };

  unsigned spikes = 0;   // 16-bit channel mask, OR-ed over all t and this lane's f

  produce(0, 0);
  __syncthreads();

  for (int tt = 0; tt < TSPAN; ++tt) {
    const int cur = tt & 1;

    // ---- produce next time-row into the other buffer (overlaps wmma) -----
    if (tt + 1 < TSPAN) produce(tt + 1, cur ^ 1);

    // ---- consume: 16 x v_wmma_f32_16x16x32_f16 in 2 groups of 8 ----------
    const v16h* bptr = ((const v16h*)Bf[cur]) + lane;   // frag kc at [kc*32]
    v8f acc = {0.f, 0.f, 0.f, 0.f, 0.f, 0.f, 0.f, 0.f};
#pragma unroll
    for (int g = 0; g < 2; ++g) {
      v16h b[8];
#pragma unroll
      for (int u = 0; u < 8; ++u)              // 16 ds_load_b128 up front
        b[u] = bptr[(g * 8 + u) * 32];
#pragma unroll
      for (int u = 0; u < 8; ++u)              // wmma chain overlaps loads
        acc = __builtin_amdgcn_wmma_f32_16x16x32_f16(
            false, afrag[g * 8 + u], false, b[u], (short)0, acc, false, false);
    }

    // ---- spike + time/band pooling: OR threshold bits ---------------------
    // C layout: VGPR v, lane L -> M = v + 8*(L>>4), N = L&15
    unsigned m8 = 0;
#pragma unroll
    for (int v = 0; v < 8; ++v)
      m8 |= (acc[v] >= THR) ? (1u << v) : 0u;
    spikes |= m8 << (p * 8);

    __syncthreads();  // buffer swap point
  }

  // ---- commit pooled spikes: one 16-bit mask per (s, ctile, fp) -----------
  atomicOr(&pool[(s * 8 + wave) * FP_OUT + fp], spikes);
}

// ---------------------------------------------------------------------------
// Kernel 3: exact integer replay of the reference k-winners reduction.
// total[c,fp] = count * (valearly + v), v = 8 iff any (count>0 && valearly).
// ---------------------------------------------------------------------------
__global__ void winner_kernel(const unsigned* __restrict__ pool, int* __restrict__ out) {
  __shared__ int s_any;
  __shared__ int s_val[N_CHANNELS];
  __shared__ int s_flat[N_CHANNELS];

  const int c = threadIdx.x;           // 128 threads, one per channel
  const int ctile = c >> 4;
  const int bit   = c & 15;

  int cnt[FP_OUT], ve[FP_OUT];
  int localAny = 0;
#pragma unroll
  for (int fp = 0; fp < FP_OUT; ++fp) {
    int mask = 0;
#pragma unroll
    for (int s = 0; s < N_SECTIONS; ++s)
      mask |= ((pool[(s * 8 + ctile) * FP_OUT + fp] >> bit) & 1) << s;
    int count = __popc(mask);
    int e = 8 - count;
    e = e < 0 ? 0 : (e > 7 ? 7 : e);
    int val = (mask >> e) & 1;
    cnt[fp] = count;
    ve[fp]  = val;
    if (count > 0 && val) localAny = 1;
  }

  if (c == 0) s_any = 0;
  __syncthreads();
  if (localAny) atomicOr(&s_any, 1);
  __syncthreads();
  const int v = s_any ? 8 : 0;

  int bv = -1, bfp = 0;
#pragma unroll
  for (int fp = 0; fp < FP_OUT; ++fp) {
    int tot = cnt[fp] * (ve[fp] + v);
    if (tot > bv) { bv = tot; bfp = fp; }
  }
  s_val[c]  = bv;
  s_flat[c] = c * FP_OUT + bfp;
  __syncthreads();

  if (c == 0) {
    int g = -1, gf = 0;
    for (int i = 0; i < N_CHANNELS; ++i)
      if (s_val[i] > g) { g = s_val[i]; gf = s_flat[i]; }
    out[0] = (g != 0) ? (gf / FP_OUT) : -1;
  }
}

// ---------------------------------------------------------------------------
// Host-side launch
// ---------------------------------------------------------------------------
extern "C" void kernel_launch(void* const* d_in, const int* in_sizes, int n_in,
                              void* d_out, int out_size, void* d_ws, size_t ws_size,
                              hipStream_t stream) {
  (void)in_sizes; (void)n_in; (void)out_size; (void)ws_size;

  const float* X = (const float*)d_in[0];   // [3231, 256] f32
  const float* W = (const float*)d_in[1];   // [8, 128, 1, 32, 16] f32

  _Float16* Xh  = (_Float16*)d_ws;
  _Float16* Wh  = Xh + XH_ELEMS;
  unsigned* pool = (unsigned*)((char*)d_ws + (size_t)(XH_ELEMS + WH_ELEMS) * sizeof(_Float16));
  int* out = (int*)d_out;

  // 1) convert to f16 + zero pool
  int cvt_blocks = (XH_ELEMS + 255) / 256;  // covers W and pool too
  cvt_kernel<<<cvt_blocks, 256, 0, stream>>>(X, W, Xh, Wh, pool);

  // 2) WMMA conv + spike + pool : grid (t-chunk, band, section)
  dim3 grid(TCHUNKS, FP_OUT, N_SECTIONS);
  conv_spike_kernel<<<grid, 256, 0, stream>>>(Xh, Wh, pool);

  // 3) winner reduction -> single int32
  winner_kernel<<<1, N_CHANNELS, 0, stream>>>(pool, out);
}
